// GNNVariableK_72499047956659
// MI455X (gfx1250) — matrix operand order repaired and checked by reading, not compiled, not run on hardware
//
#include <hip/hip_runtime.h>
#include <hip/hip_bf16.h>
#include <math.h>

typedef __attribute__((ext_vector_type(16))) _Float16 v16h;
typedef __attribute__((ext_vector_type(8)))  float    v8f;

#define NNODES 100000
#define KDEG   24
#define NPW    16                 // nodes per workgroup
#define EPW    (NPW * KDEG)       // 384 edges per workgroup
#define NTILES (EPW / 16)         // 24 wmma row-tiles
#define BLOCK  256
#define NWAVES (BLOCK / 32)

__device__ __forceinline__ float angle3(float ax, float ay, float az,
                                        float bx, float by, float bz) {
    float cx = ay * bz - az * by;
    float cy = az * bx - ax * bz;
    float cz = ax * by - ay * bx;
    float cn = sqrtf(cx * cx + cy * cy + cz * cz + 1e-12f);
    float d  = ax * bx + ay * by + az * bz;
    return atan2f(cn, d);
}

// One 16-row tile through: relu(A(16x32) @ W1(32x32) + b1) @ W2(32x16) + b2
// A is a pre-assembled wmma A-fragment (K padded to 32).
// w1T/w2T are f16 weights stored transposed+padded: wT[out*32 + k].
__device__ __forceinline__ void stage_gemm_32_16(
    v16h a,
    const _Float16* __restrict__ w1T, const float* __restrict__ b1,
    const _Float16* __restrict__ w2T, const float* __restrict__ b2,
    _Float16* sHw,        // per-wave 16x32 f16 transpose scratch
    _Float16* xout,       // LDS out tile base: [16 rows][16 cols] f16
    int lane)
{
    const int col = lane & 15;
    const int hi  = lane >> 4;      // half-select
    // hidden(16x32) = A @ W1 : two 16-col tiles
    #pragma unroll
    for (int t = 0; t < 2; ++t) {
        v16h b;
        #pragma unroll
        for (int i = 0; i < 8; ++i) {
            b[i]     = w1T[(t * 16 + col) * 32 + hi * 8 + i];
            b[8 + i] = w1T[(t * 16 + col) * 32 + 16 + hi * 8 + i];
        }
        v8f c = {};
        c = __builtin_amdgcn_wmma_f32_16x16x32_f16(false, a, false, b,
                                                   (short)0, c, false, false);
        float bias = b1[t * 16 + col];
        #pragma unroll
        for (int r = 0; r < 8; ++r) {
            float v = c[r] + bias;
            sHw[(r + hi * 8) * 32 + t * 16 + col] =
                (_Float16)(v > 0.0f ? v : 0.0f);
        }
    }
    __builtin_amdgcn_wave_barrier();   // DS is in-order within a wave
    // re-load hidden in A-fragment layout
    const int kb0 = hi * 8;
    v16h a2;
    {
        const _Float16* p = sHw + col * 32 + kb0;
        #pragma unroll
        for (int i = 0; i < 8; ++i) a2[i] = p[i];
        #pragma unroll
        for (int i = 0; i < 8; ++i) a2[8 + i] = p[16 + i];
    }
    v16h b2v;
    #pragma unroll
    for (int i = 0; i < 8; ++i) {
        b2v[i]     = w2T[col * 32 + hi * 8 + i];
        b2v[8 + i] = w2T[col * 32 + 16 + hi * 8 + i];
    }
    v8f d = {};
    d = __builtin_amdgcn_wmma_f32_16x16x32_f16(false, a2, false, b2v,
                                               (short)0, d, false, false);
    float bias2 = b2[col];
    #pragma unroll
    for (int r = 0; r < 8; ++r)
        xout[(r + hi * 8) * 16 + col] = (_Float16)(d[r] + bias2);
    __builtin_amdgcn_wave_barrier();
}

// small per-node MLP on one thread (fh == 32)
__device__ __forceinline__ void node_mlp(
    const float* in, int fin,
    const float* __restrict__ w1, const float* __restrict__ b1,
    const float* __restrict__ w2, const float* __restrict__ b2,
    int fo, float* out)
{
    float h[32];
    for (int o = 0; o < 32; ++o) {
        float s = b1[o];
        for (int k = 0; k < fin; ++k) s += in[k] * w1[k * 32 + o];
        h[o] = s > 0.0f ? s : 0.0f;
    }
    for (int o = 0; o < fo; ++o) {
        float s = b2[o];
        for (int k = 0; k < 32; ++k) s += h[k] * w2[k * fo + o];
        out[o] = s;
    }
}

__device__ __forceinline__ void scatter_mean16(const _Float16* sX,
                                               float (*sNode)[32], int tid) {
    const int node = tid >> 4, f = tid & 15;
    const _Float16* p = sX + (node * KDEG) * 16 + f;
    float s = 0.0f;
    #pragma unroll
    for (int j = 0; j < KDEG; ++j) s += (float)p[j * 16];
    sNode[node][f] = s * (1.0f / (float)KDEG);
}

__global__ __launch_bounds__(BLOCK, 2)
void gnn_fused_kernel(
    const float* __restrict__ pos, const float* __restrict__ old_w,
    const float* __restrict__ normals, const float* __restrict__ stddev,
    const int* __restrict__ cols,
    const float* __restrict__ l1_w1, const float* __restrict__ l1_b1,
    const float* __restrict__ l1_w2, const float* __restrict__ l1_b2,
    const float* __restrict__ g1_w1, const float* __restrict__ g1_b1,
    const float* __restrict__ g1_w2, const float* __restrict__ g1_b2,
    const float* __restrict__ l2_w1, const float* __restrict__ l2_b1,
    const float* __restrict__ l2_w2, const float* __restrict__ l2_b2,
    const float* __restrict__ g2_w1, const float* __restrict__ g2_b1,
    const float* __restrict__ g2_w2, const float* __restrict__ g2_b2,
    const float* __restrict__ l3_w1, const float* __restrict__ l3_b1,
    const float* __restrict__ l3_w2, const float* __restrict__ l3_b2,
    const float* __restrict__ g3_w1, const float* __restrict__ g3_b1,
    const float* __restrict__ g3_w2, const float* __restrict__ g3_b2,
    const float* __restrict__ l4_w1, const float* __restrict__ l4_b1,
    const float* __restrict__ l4_w2, const float* __restrict__ l4_b2,
    float* __restrict__ out)
{
    // f16 weights, transposed + K-padded to 32: wT[o*32 + k]
    __shared__ __attribute__((aligned(32))) _Float16 sW1a[32 * 32];
    __shared__ __attribute__((aligned(32))) _Float16 sW1b[16 * 32];
    __shared__ __attribute__((aligned(32))) _Float16 sW2a[32 * 32];
    __shared__ __attribute__((aligned(32))) _Float16 sW2b[16 * 32];
    __shared__ __attribute__((aligned(32))) _Float16 sW3a[32 * 32];
    __shared__ __attribute__((aligned(32))) _Float16 sW3b[16 * 32];
    __shared__ __attribute__((aligned(32))) _Float16 sW4a[64 * 32];
    __shared__ float sB1a[32], sB1b[16], sB2a[32], sB2b[16];
    __shared__ float sB3a[32], sB3b[16], sB4a[64], sW4b[64];
    __shared__ _Float16 sF[EPW * 8];        // stage-1 edge features
    __shared__ _Float16 sX[EPW * 16];       // edge hidden (reused each stage)
    __shared__ float    sCart[EPW * 3];
    __shared__ float    sNode[NPW][32];     // means / xg / mat / rest / denom
    __shared__ _Float16 sH[NWAVES][16 * 32];
    __shared__ float    sLogit[EPW];

    const int tid  = threadIdx.x;
    const int wave = tid >> 5;
    const int lane = tid & 31;
    const int wg   = blockIdx.x;
    const int nodeBase = wg * NPW;
    const long long edgeBase = (long long)wg * EPW;

    // ---- preload + transpose/pad weights to f16 --------------------------
    {
        auto loadT = [&](_Float16* dst, const float* w, int fin, int fout) {
            for (int i = tid; i < fout * 32; i += BLOCK) {
                int o = i >> 5, k = i & 31;
                dst[i] = (k < fin) ? (_Float16)w[k * fout + o] : (_Float16)0.0f;
            }
        };
        loadT(sW1a, l1_w1,  8, 32); loadT(sW1b, l1_w2, 32, 16);
        loadT(sW2a, l2_w1, 24, 32); loadT(sW2b, l2_w2, 32, 16);
        loadT(sW3a, l3_w1, 24, 32); loadT(sW3b, l3_w2, 32, 16);
        loadT(sW4a, l4_w1, 27, 64);
        for (int i = tid; i < 32; i += BLOCK) { sB1a[i]=l1_b1[i]; sB2a[i]=l2_b1[i]; sB3a[i]=l3_b1[i]; }
        for (int i = tid; i < 16; i += BLOCK) { sB1b[i]=l1_b2[i]; sB2b[i]=l2_b2[i]; sB3b[i]=l3_b2[i]; }
        for (int i = tid; i < 64; i += BLOCK) { sB4a[i]=l4_b1[i]; sW4b[i]=l4_w2[i]; }
    }
    __syncthreads();

    // ---- stage 0: edge geometric features --------------------------------
    {
        const float scale = 0.2f / stddev[0];
        for (int e = tid; e < EPW; e += BLOCK) {
            const int nG = nodeBase + e / KDEG;
            const int c  = cols[edgeBase + e];
            const float px = pos[nG*3+0], py = pos[nG*3+1], pz = pos[nG*3+2];
            const float dx = (pos[c*3+0] - px) * scale;
            const float dy = (pos[c*3+1] - py) * scale;
            const float dz = (pos[c*3+2] - pz) * scale;
            sCart[e*3+0] = dx; sCart[e*3+1] = dy; sCart[e*3+2] = dz;
            const float w   = old_w[edgeBase + e];
            const float dst = sqrtf(dx*dx + dy*dy + dz*dz + 1e-12f);
            const float ax = normals[nG*3+0], ay = normals[nG*3+1], az = normals[nG*3+2];
            const float bx = normals[c*3+0],  by = normals[c*3+1],  bz = normals[c*3+2];
            _Float16* f = sF + e * 8;
            f[0]=(_Float16)dx; f[1]=(_Float16)dy; f[2]=(_Float16)dz; f[3]=(_Float16)w;
            f[4]=(_Float16)dst;
            f[5]=(_Float16)angle3(ax,ay,az, dx,dy,dz);
            f[6]=(_Float16)angle3(bx,by,bz, dx,dy,dz);
            f[7]=(_Float16)angle3(ax,ay,az, bx,by,bz);
        }
    }
    __syncthreads();

    _Float16* myH = &sH[wave][0];

    // ---- stage l1: 8 -> 32 -> 16 via WMMA --------------------------------
    for (int t = wave; t < NTILES; t += NWAVES) {
        const int row = lane & 15;
        v16h a = {};
        if (lane < 16) {    // K0..7 live in the low half-lanes only
            const _Float16* p = sF + (t * 16 + row) * 8;
            #pragma unroll
            for (int i = 0; i < 8; ++i) a[i] = p[i];
        }
        stage_gemm_32_16(a, sW1a, sB1a, sW1b, sB1b, myH, sX + t * 256, lane);
    }
    __syncthreads();

    // ---- g1: mean(16) ++ normals(3) -> 32 -> 8 ---------------------------
    scatter_mean16(sX, sNode, tid);
    __syncthreads();
    if (tid < NPW) {
        float gin[19];
        #pragma unroll
        for (int k = 0; k < 16; ++k) gin[k] = sNode[tid][k];
        const int nG = nodeBase + tid;
        gin[16] = normals[nG*3+0]; gin[17] = normals[nG*3+1]; gin[18] = normals[nG*3+2];
        float xg[8];
        node_mlp(gin, 19, g1_w1, g1_b1, g1_w2, g1_b2, 8, xg);
        #pragma unroll
        for (int o = 0; o < 8; ++o) sNode[tid][o] = xg[o];
    }
    __syncthreads();

    // ---- stage l2: [x(16) ++ xg(8)] -> 32 -> 16 --------------------------
    for (int t = wave; t < NTILES; t += NWAVES) {
        const int row = lane & 15, edge = t * 16 + row, node = edge / KDEG;
        const int hi = lane >> 4, kb0 = hi * 8;
        v16h a;
        const _Float16* xp = sX + edge * 16;
        #pragma unroll
        for (int i = 0; i < 8; ++i) a[i] = xp[kb0 + i];
        if (hi == 0) {
            #pragma unroll
            for (int i = 0; i < 8; ++i) a[8 + i] = (_Float16)sNode[node][i];
        } else {
            #pragma unroll
            for (int i = 0; i < 8; ++i) a[8 + i] = (_Float16)0.0f;
        }
        stage_gemm_32_16(a, sW2a, sB2a, sW2b, sB2b, myH, sX + t * 256, lane);
    }
    __syncthreads();

    // ---- g2 --------------------------------------------------------------
    scatter_mean16(sX, sNode, tid);
    __syncthreads();
    if (tid < NPW) {
        float gin[16], xg[8];
        #pragma unroll
        for (int k = 0; k < 16; ++k) gin[k] = sNode[tid][k];
        node_mlp(gin, 16, g2_w1, g2_b1, g2_w2, g2_b2, 8, xg);
        #pragma unroll
        for (int o = 0; o < 8; ++o) sNode[tid][o] = xg[o];
    }
    __syncthreads();

    // ---- stage l3 --------------------------------------------------------
    for (int t = wave; t < NTILES; t += NWAVES) {
        const int row = lane & 15, edge = t * 16 + row, node = edge / KDEG;
        const int hi = lane >> 4, kb0 = hi * 8;
        v16h a;
        const _Float16* xp = sX + edge * 16;
        #pragma unroll
        for (int i = 0; i < 8; ++i) a[i] = xp[kb0 + i];
        if (hi == 0) {
            #pragma unroll
            for (int i = 0; i < 8; ++i) a[8 + i] = (_Float16)sNode[node][i];
        } else {
            #pragma unroll
            for (int i = 0; i < 8; ++i) a[8 + i] = (_Float16)0.0f;
        }
        stage_gemm_32_16(a, sW3a, sB3a, sW3b, sB3b, myH, sX + t * 256, lane);
    }
    __syncthreads();

    // ---- g3 + quaternion -> rotation matrix ------------------------------
    scatter_mean16(sX, sNode, tid);
    __syncthreads();
    if (tid < NPW) {
        float gin[16], xg[12];
        #pragma unroll
        for (int k = 0; k < 16; ++k) gin[k] = sNode[tid][k];
        node_mlp(gin, 16, g3_w1, g3_b1, g3_w2, g3_b2, 12, xg);
        float qn = sqrtf(xg[0]*xg[0] + xg[1]*xg[1] + xg[2]*xg[2] + xg[3]*xg[3]) + 1e-8f;
        float w = xg[0]/qn, x = xg[1]/qn, y = xg[2]/qn, z = xg[3]/qn;
        float* m = &sNode[tid][8];
        m[0] = 1.f - 2.f*(y*y + z*z); m[1] = 2.f*(x*y - z*w); m[2] = 2.f*(x*z + y*w);
        m[3] = 2.f*(x*y + z*w); m[4] = 1.f - 2.f*(x*x + z*z); m[5] = 2.f*(y*z - x*w);
        m[6] = 2.f*(x*z - y*w); m[7] = 2.f*(y*z + x*w); m[8] = 1.f - 2.f*(x*x + y*y);
        #pragma unroll
        for (int o = 0; o < 8; ++o) sNode[tid][17 + o] = xg[4 + o];
    }
    __syncthreads();

    // ---- stage l4: [x(16) ++ rest(8) ++ rot_cart(3)] -> 64 -> 1 ----------
    const float b2s = l4_b2[0];
    for (int t = wave; t < NTILES; t += NWAVES) {
        const int row = lane & 15, edge = t * 16 + row, node = edge / KDEG;
        const int hi = lane >> 4, col = lane & 15, kb0 = hi * 8;
        v16h a;
        const _Float16* xp = sX + edge * 16;
        #pragma unroll
        for (int i = 0; i < 8; ++i) a[i] = xp[kb0 + i];
        if (hi == 0) {          // K16..23 = rest
            #pragma unroll
            for (int i = 0; i < 8; ++i) a[8 + i] = (_Float16)sNode[node][17 + i];
        } else {                // K24..26 = rot_cart, K27..31 = 0
            const float cx = sCart[edge*3+0], cy = sCart[edge*3+1], cz = sCart[edge*3+2];
            const float* m = &sNode[node][8];
            a[8]  = (_Float16)(m[0]*cx + m[1]*cy + m[2]*cz);
            a[9]  = (_Float16)(m[3]*cx + m[4]*cy + m[5]*cz);
            a[10] = (_Float16)(m[6]*cx + m[7]*cy + m[8]*cz);
            #pragma unroll
            for (int i = 3; i < 8; ++i) a[8 + i] = (_Float16)0.0f;
        }
        float acc[8];
        #pragma unroll
        for (int r = 0; r < 8; ++r) acc[r] = 0.0f;
        #pragma unroll
        for (int nt = 0; nt < 4; ++nt) {
            v16h b;
            #pragma unroll
            for (int i = 0; i < 8; ++i) {
                b[i]     = sW4a[(nt * 16 + col) * 32 + hi * 8 + i];
                b[8 + i] = sW4a[(nt * 16 + col) * 32 + 16 + hi * 8 + i];
            }
            v8f c = {};
            c = __builtin_amdgcn_wmma_f32_16x16x32_f16(false, a, false, b,
                                                       (short)0, c, false, false);
            const float bias = sB4a[nt * 16 + col];
            const float w2c  = sW4b[nt * 16 + col];
            #pragma unroll
            for (int r = 0; r < 8; ++r) {
                float v = c[r] + bias;
                acc[r] += (v > 0.0f ? v : 0.0f) * w2c;
            }
        }
        #pragma unroll
        for (int r = 0; r < 8; ++r) {
            #pragma unroll
            for (int msk = 1; msk < 16; msk <<= 1)
                acc[r] += __shfl_xor(acc[r], msk, 32);
        }
        if (col == 0) {
            #pragma unroll
            for (int r = 0; r < 8; ++r)
                sLogit[t * 16 + r + hi * 8] = acc[r] + b2s;
        }
    }
    __syncthreads();

    // ---- segment softmax over each node's 24 edges -----------------------
    for (int e = tid; e < EPW; e += BLOCK) sLogit[e] = __expf(sLogit[e]);
    __syncthreads();
    if (tid < NPW) {
        float s = 0.0f;
        #pragma unroll
        for (int j = 0; j < KDEG; ++j) s += sLogit[tid * KDEG + j];
        sNode[tid][25] = s;
    }
    __syncthreads();
    for (int e = tid; e < EPW; e += BLOCK)
        out[edgeBase + e] = sLogit[e] / sNode[e / KDEG][25];
}

extern "C" void kernel_launch(void* const* d_in, const int* in_sizes, int n_in,
                              void* d_out, int out_size, void* d_ws, size_t ws_size,
                              hipStream_t stream) {
    (void)in_sizes; (void)n_in; (void)out_size; (void)d_ws; (void)ws_size;
    const float* pos     = (const float*)d_in[0];
    const float* old_w   = (const float*)d_in[1];
    const float* normals = (const float*)d_in[2];
    const float* stddev  = (const float*)d_in[3];
    const int*   cols    = ((const int*)d_in[4]) + (long long)NNODES * KDEG; // [rows; cols]

    dim3 grid(NNODES / NPW);   // 6250
    dim3 block(BLOCK);
    gnn_fused_kernel<<<grid, block, 0, stream>>>(
        pos, old_w, normals, stddev, cols,
        (const float*)d_in[5],  (const float*)d_in[6],  (const float*)d_in[7],  (const float*)d_in[8],
        (const float*)d_in[9],  (const float*)d_in[10], (const float*)d_in[11], (const float*)d_in[12],
        (const float*)d_in[13], (const float*)d_in[14], (const float*)d_in[15], (const float*)d_in[16],
        (const float*)d_in[17], (const float*)d_in[18], (const float*)d_in[19], (const float*)d_in[20],
        (const float*)d_in[21], (const float*)d_in[22], (const float*)d_in[23], (const float*)d_in[24],
        (const float*)d_in[25], (const float*)d_in[26], (const float*)d_in[27], (const float*)d_in[28],
        (const float*)d_in[29], (const float*)d_in[30], (const float*)d_in[31], (const float*)d_in[32],
        (float*)d_out);
}
